// VolumetricRayMarcher_72473278152919
// MI455X (gfx1250) — compile-verified, matrix operand order checked by reading, not run on hardware
//
#include <hip/hip_runtime.h>

// ---------------------------------------------------------------------------
// Volumetric ray marcher (NeRF-style) for MI455X / gfx1250.
//   points -> h = relu(points @ W1)            [K=3 -> 32]   (2x WMMA f16)
//   h      -> [rgb_logits, density_logit]      [K=32 -> 4]   (1x WMMA f16)
//   alpha compositing scan over S=128 samples  (VALU, sequential per ray)
// One wave owns 16 rays (M dim of the 16x16x32 WMMA tiles). LDS is used only
// for intra-wave layout transposes (C/D layout -> A layout), relying on the
// CDNA5 guarantee that DS ops from one wave execute in order (s_wait_dscnt
// handles the VGPR dependency).
// ---------------------------------------------------------------------------

typedef _Float16 v16h __attribute__((ext_vector_type(16)));
typedef float    v8f  __attribute__((ext_vector_type(8)));

#define S_SAMPLES 128
#define T_NEAR 0.1f
#define T_FAR  2.0f
#define WAVES_PER_BLOCK 4
#define RAYS_PER_WAVE 16
#define RAYS_PER_BLOCK (WAVES_PER_BLOCK * RAYS_PER_WAVE)

// K-pair base held by VGPR j of a 16-bit A/B operand (ISA 7.12.2):
//   VGPR0..3 : K = {0,1},{2,3},{4,5},{6,7}   (+8 for lanes 16..31)
//   VGPR4..7 : K = {16,17},...,{22,23}       (+8 for lanes 16..31)
__device__ __forceinline__ int kpair_base(int j, int half) {
    return (j < 4) ? (2 * j + 8 * half) : (16 + 2 * (j - 4) + 8 * half);
}

__global__ __launch_bounds__(WAVES_PER_BLOCK * 32)
void VolumetricRayMarcher_72473278152919_kernel(
    const float* __restrict__ ro,    // [B*R, 3]
    const float* __restrict__ rd,    // [B*R, 3]
    const float* __restrict__ W1,    // [3, 32]
    const float* __restrict__ Wrgb,  // [32, 3]
    const float* __restrict__ Wd,    // [32, 1]
    float* __restrict__ out,         // [B*R, 3]
    int nRays)
{
    // Per-wave LDS slices: hbuf = 16x32 f32 (hidden), obuf = 16x16 f32 (layer2 D)
    __shared__ float smem[WAVES_PER_BLOCK][16 * 32 + 16 * 16];

    const int lane  = threadIdx.x & 31;
    const int wave  = threadIdx.x >> 5;
    const int row   = lane & 15;   // ray row (M) / column (N) index
    const int half  = lane >> 4;   // 0: lanes 0-15, 1: lanes 16-31
    const int waveId = blockIdx.x * WAVES_PER_BLOCK + wave;
    const int ray    = waveId * RAYS_PER_WAVE + row;
    const int rayC   = (ray < nRays) ? ray : (nRays - 1);

    float* hbuf = &smem[wave][0];
    float* obuf = &smem[wave][16 * 32];

    // Ray origin/direction for this lane's row (duplicated across halves).
    const float ox = ro[rayC * 3 + 0], oy = ro[rayC * 3 + 1], oz = ro[rayC * 3 + 2];
    const float dx = rd[rayC * 3 + 0], dy = rd[rayC * 3 + 1], dz = rd[rayC * 3 + 2];

    // ---- Build constant B operands (weights) in WMMA B layout -------------
    // B1 (32x16 f16, two tiles for hidden units 0-15 / 16-31): only K=0..2
    // nonzero -> only lanes 0-15 (K 0,1 in VGPR0; K 2 in VGPR1) carry data.
    v16h b1lo = {}, b1hi = {}, b2 = {};
    if (half == 0) {
        b1lo[0] = (_Float16)W1[0 * 32 + row];
        b1lo[1] = (_Float16)W1[1 * 32 + row];
        b1lo[2] = (_Float16)W1[2 * 32 + row];
        b1hi[0] = (_Float16)W1[0 * 32 + 16 + row];
        b1hi[1] = (_Float16)W1[1 * 32 + 16 + row];
        b1hi[2] = (_Float16)W1[2 * 32 + 16 + row];
    }
    // B2 (32x16 f16): cols 0..2 = W_rgb, col 3 = W_d, cols 4..15 = 0.
#pragma unroll
    for (int j = 0; j < 8; ++j) {
        const int k0 = kpair_base(j, half);
        float w0 = 0.0f, w1 = 0.0f;
        if (row < 3)       { w0 = Wrgb[k0 * 3 + row]; w1 = Wrgb[(k0 + 1) * 3 + row]; }
        else if (row == 3) { w0 = Wd[k0];             w1 = Wd[k0 + 1]; }
        b2[2 * j]     = (_Float16)w0;
        b2[2 * j + 1] = (_Float16)w1;
    }

    const float dt = (T_FAR - T_NEAR) / (float)(S_SAMPLES - 1);
    float T = 1.0f;                     // transmittance (exclusive cumprod)
    float accR = 0.0f, accG = 0.0f, accB = 0.0f;

    for (int s = 0; s < S_SAMPLES; ++s) {
        const float t = T_NEAR + dt * (float)s;

        // ---- A1: 16 rays x 32 (K=0..2 = point coords) ---------------------
        v16h a1 = {};
        if (half == 0) {
            a1[0] = (_Float16)(ox + dx * t);
            a1[1] = (_Float16)(oy + dy * t);
            a1[2] = (_Float16)(oz + dz * t);
        }

        const v8f cz = {};
        v8f h0 = __builtin_amdgcn_wmma_f32_16x16x32_f16(false, a1, false, b1lo,
                                                        (short)0, cz, false, false);
        v8f h1 = __builtin_amdgcn_wmma_f32_16x16x32_f16(false, a1, false, b1hi,
                                                        (short)0, cz, false, false);

        // ReLU + transpose to LDS: hbuf[m][k], m = ray row, k = hidden unit.
        // C/D layout: lane holds column n=row, VGPR j holds row m=j+8*half.
#pragma unroll
        for (int j = 0; j < 8; ++j) {
            const int m = j + 8 * half;
            hbuf[m * 32 + row]      = fmaxf(h0[j], 0.0f);
            hbuf[m * 32 + 16 + row] = fmaxf(h1[j], 0.0f);
        }
        __asm__ volatile("s_wait_dscnt 0" ::: "memory");

        // ---- A2: this lane's ray row of h, packed in A layout -------------
        v16h a2;
#pragma unroll
        for (int j = 0; j < 8; ++j) {
            const int k0 = kpair_base(j, half);
            a2[2 * j]     = (_Float16)hbuf[row * 32 + k0];
            a2[2 * j + 1] = (_Float16)hbuf[row * 32 + k0 + 1];
        }

        v8f d2 = __builtin_amdgcn_wmma_f32_16x16x32_f16(false, a2, false, b2,
                                                        (short)0, cz, false, false);

        // Transpose D back to lane-per-ray via LDS.
#pragma unroll
        for (int j = 0; j < 8; ++j)
            obuf[(j + 8 * half) * 16 + row] = d2[j];
        __asm__ volatile("s_wait_dscnt 0" ::: "memory");

        const float pr = obuf[row * 16 + 0];
        const float pg = obuf[row * 16 + 1];
        const float pb = obuf[row * 16 + 2];
        const float pd = obuf[row * 16 + 3];

        // ---- Compositing update (matches reference exactly) ---------------
        const float density = fmaxf(pd, 0.0f);
        const float delta   = (s == S_SAMPLES - 1) ? 1e10f : dt;
        const float alpha   = 1.0f - __expf(-density * delta);
        const float w       = alpha * T;
        accR += w * (1.0f / (1.0f + __expf(-pr)));
        accG += w * (1.0f / (1.0f + __expf(-pg)));
        accB += w * (1.0f / (1.0f + __expf(-pb)));
        T *= (1.0f - alpha + 1e-10f);
    }

    // Lanes 0-15 own the 16 rays of this wave; lanes 16-31 were duplicates.
    if (half == 0 && ray < nRays) {
        out[ray * 3 + 0] = accR;
        out[ray * 3 + 1] = accG;
        out[ray * 3 + 2] = accB;
    }
}

extern "C" void kernel_launch(void* const* d_in, const int* in_sizes, int n_in,
                              void* d_out, int out_size, void* d_ws, size_t ws_size,
                              hipStream_t stream) {
    const float* ro   = (const float*)d_in[0];  // ray_origins  [B,R,3]
    const float* rd   = (const float*)d_in[1];  // ray_directions [B,R,3]
    const float* W1   = (const float*)d_in[2];  // [3,32]
    const float* Wrgb = (const float*)d_in[3];  // [32,3]
    const float* Wd   = (const float*)d_in[4];  // [32,1]
    float* out        = (float*)d_out;          // [B,R,3]

    const int nRays  = in_sizes[0] / 3;                       // 16384
    const int blocks = (nRays + RAYS_PER_BLOCK - 1) / RAYS_PER_BLOCK;  // 256

    hipLaunchKernelGGL(VolumetricRayMarcher_72473278152919_kernel,
                       dim3(blocks), dim3(WAVES_PER_BLOCK * 32), 0, stream,
                       ro, rd, W1, Wrgb, Wd, out, nRays);
}
